// SparseMoE_53369263620289
// MI455X (gfx1250) — matrix operand order repaired and checked by reading, not compiled, not run on hardware
//
#include <hip/hip_runtime.h>
#include <hip/hip_bf16.h>

// Problem constants (from reference)
#define TK_B 4
#define TK_S 4096
#define TK_T (TK_B * TK_S)   // 16384 tokens
#define TK_D 512
#define TK_E 8
#define TK_H 2048
#define TILE_M 48            // 3 WMMA row-tiles share every B fragment
#define EXPERT_GRID 690      // floor(2*T/48) + E

// dynamic LDS layout (bytes)
#define XS_STRIDE 520        // 512 + 8 pad (bank rotate)
#define HS_STRIDE 264        // 256 + 8 pad
#define SMEM_XS    0
#define SMEM_HS    (TILE_M * XS_STRIDE * 2)                  // 49920
#define SMEM_GS    (SMEM_HS + TILE_M * HS_STRIDE * 2)        // 75264
#define SMEM_TOK   (SMEM_GS + TILE_M * 4)                    // 75456
#define SMEM_BYTES (SMEM_TOK + TILE_M * 4 + 128)             // ~75.8 KB (<320KB WGP LDS)

typedef __attribute__((ext_vector_type(16))) __bf16 v16bf;
typedef __attribute__((ext_vector_type(8)))  __bf16 v8bf;
typedef __attribute__((ext_vector_type(8)))  float  v8f;

// float -> bf16 (round to nearest even), as raw u16
__device__ __forceinline__ unsigned short f2bf_raw(float f) {
    unsigned u = __builtin_bit_cast(unsigned, f);
    return (unsigned short)((u + 0x7FFFu + ((u >> 16) & 1u)) >> 16);
}

__device__ __forceinline__ unsigned hash_u32(unsigned x) {
    x ^= x >> 17; x *= 0xED5AD4BBu;
    x ^= x >> 11; x *= 0xAC4C1B51u;
    x ^= x >> 15; x *= 0x31848BABu;
    x ^= x >> 14;
    return x;
}

// ---------------------------------------------------------------------------
// 0) zero output + expert counters
// ---------------------------------------------------------------------------
__global__ void moe_zero_kernel(float* __restrict__ out, int* __restrict__ cnt, int n) {
    int i = blockIdx.x * blockDim.x + threadIdx.x;
    if (i < TK_E) cnt[i] = 0;
    for (; i < n; i += gridDim.x * blockDim.x) out[i] = 0.0f;
}

// ---------------------------------------------------------------------------
// 1) convert x (f32) -> bf16 ; transpose+convert weights
// ---------------------------------------------------------------------------
__global__ void moe_cvt_x_kernel(const float* __restrict__ x,
                                 unsigned short* __restrict__ xbf, int n) {
    for (int i = blockIdx.x * blockDim.x + threadIdx.x; i < n;
         i += gridDim.x * blockDim.x)
        xbf[i] = f2bf_raw(x[i]);
}

// W1 [E][D][H] f32 -> w1t [E][H][D] bf16
__global__ void moe_tw1_kernel(const float* __restrict__ W1,
                               unsigned short* __restrict__ w1t, int n) {
    for (int i = blockIdx.x * blockDim.x + threadIdx.x; i < n;
         i += gridDim.x * blockDim.x) {
        int e   = i / (TK_H * TK_D);
        int rem = i - e * (TK_H * TK_D);
        int h   = rem / TK_D;
        int d   = rem - h * TK_D;
        w1t[i] = f2bf_raw(W1[((size_t)e * TK_D + d) * TK_H + h]);
    }
}

// W2 [E][H][D] f32 -> w2t [E][D][H] bf16
__global__ void moe_tw2_kernel(const float* __restrict__ W2,
                               unsigned short* __restrict__ w2t, int n) {
    for (int i = blockIdx.x * blockDim.x + threadIdx.x; i < n;
         i += gridDim.x * blockDim.x) {
        int e   = i / (TK_D * TK_H);
        int rem = i - e * (TK_D * TK_H);
        int d   = rem / TK_H;
        int h   = rem - d * TK_H;
        w2t[i] = f2bf_raw(W2[((size_t)e * TK_H + h) * TK_D + d]);
    }
}

// ---------------------------------------------------------------------------
// 2) router: logits, noisy top-2, softmax gates, append to per-expert lists
// ---------------------------------------------------------------------------
__global__ __launch_bounds__(256)
void moe_router_kernel(const float* __restrict__ x,
                       const float* __restrict__ Wr, const float* __restrict__ br,
                       const float* __restrict__ Wn, const float* __restrict__ bn,
                       int* __restrict__ cnt,
                       int* __restrict__ tokList,     // [E][T]
                       float* __restrict__ gateList)  // [E][T]
{
    __shared__ float WrS[TK_D * TK_E];   // 16 KB
    __shared__ float WnS[TK_D * TK_E];   // 16 KB
    for (int i = threadIdx.x; i < TK_D * TK_E; i += 256) {
        WrS[i] = Wr[i];
        WnS[i] = Wn[i];
    }
    __syncthreads();

    int t = blockIdx.x * 256 + threadIdx.x;   // grid sized exactly T/256
    float r[TK_E], nz[TK_E];
#pragma unroll
    for (int e = 0; e < TK_E; ++e) { r[e] = br[e]; nz[e] = bn[e]; }

    const float* xr = x + (size_t)t * TK_D;
    for (int d = 0; d < TK_D; ++d) {
        float xv = xr[d];
#pragma unroll
        for (int e = 0; e < TK_E; ++e) {
            r[e]  += xv * WrS[d * TK_E + e];
            nz[e] += xv * WnS[d * TK_E + e];
        }
    }

    float nl[TK_E];
#pragma unroll
    for (int e = 0; e < TK_E; ++e) {
        unsigned s  = (unsigned)(t * TK_E + e);
        float u1 = (hash_u32(2u * s + 1u) + 0.5f) * 2.3283064e-10f; // (0,1)
        float u2 = (hash_u32(2u * s + 2u) + 0.5f) * 2.3283064e-10f;
        float z  = sqrtf(-2.0f * logf(u1)) * cosf(6.2831853f * u2);
        float v  = nz[e];
        float sp = (v > 20.0f) ? v : log1pf(expf(v));
        nl[e] = r[e] + z * sp;
    }

    // top-2 of 8
    int i0 = 0;
#pragma unroll
    for (int e = 1; e < TK_E; ++e) if (nl[e] > nl[i0]) i0 = e;
    int i1 = (i0 == 0) ? 1 : 0;
#pragma unroll
    for (int e = 0; e < TK_E; ++e) if (e != i0 && nl[e] > nl[i1]) i1 = e;

    float m  = nl[i0];
    float e1 = expf(nl[i1] - m);
    float s  = 1.0f + e1;
    float g0 = 1.0f / s;
    float g1 = e1 / s;

    int p0 = atomicAdd(&cnt[i0], 1);
    tokList[i0 * TK_T + p0]  = t;
    gateList[i0 * TK_T + p0] = g0;
    int p1 = atomicAdd(&cnt[i1], 1);
    tokList[i1 * TK_T + p1]  = t;
    gateList[i1 * TK_T + p1] = g1;
}

// ---------------------------------------------------------------------------
// 3) scheduler: flatten per-expert counts into 48-token tile descriptors
// ---------------------------------------------------------------------------
__global__ void moe_sched_kernel(const int* __restrict__ cnt,
                                 int* __restrict__ tileDesc,
                                 int* __restrict__ numTiles) {
    if (blockIdx.x == 0 && threadIdx.x == 0) {
        int n = 0;
        for (int e = 0; e < TK_E; ++e) {
            int c = cnt[e];
            int tiles = (c + TILE_M - 1) / TILE_M;
            for (int i = 0; i < tiles; ++i)
                tileDesc[n++] = (e << 24) | (i * TILE_M);
        }
        numTiles[0] = n;
    }
}

// ---------------------------------------------------------------------------
// 4) expert FFN: 48-token tiles (3 WMMA row-tiles share every B fragment ->
//    1/3 the L2 weight traffic of M=16).  Uses CDNA5's big WGP LDS (~76 KB
//    dynamic, >64KB static cap).  k-outer / n-inner nest keeps A-fragment
//    live ranges short; H processed in 8 chunks of 256 with persistent GEMM2
//    accumulators; X rows staged via gfx1250 async-to-LDS loads.
// ---------------------------------------------------------------------------
__global__ __launch_bounds__(256)
void moe_expert_kernel(const unsigned short* __restrict__ xbf,   // [T][D]
                       const unsigned short* __restrict__ w1t,   // [E][H][D]
                       const unsigned short* __restrict__ w2t,   // [E][D][H]
                       const float* __restrict__ b1,             // [E][H]
                       const float* __restrict__ b2,             // [E][D]
                       const int* __restrict__ tokList,          // [E][T]
                       const float* __restrict__ gateList,       // [E][T]
                       const int* __restrict__ cnt,              // [E]
                       const int* __restrict__ tileDesc,
                       const int* __restrict__ numTiles,
                       float* __restrict__ out)                  // [T][D]
{
    extern __shared__ char smem[];
    unsigned short* Xs = (unsigned short*)(smem + SMEM_XS);  // [48][520]
    unsigned short* Hs = (unsigned short*)(smem + SMEM_HS);  // [48][264]
    float*          gS = (float*)(smem + SMEM_GS);           // [48]
    int*          tokS = (int*)(smem + SMEM_TOK);            // [48]

    int bid = blockIdx.x;
    if (bid >= numTiles[0]) return;              // uniform per block

    int desc  = tileDesc[bid];
    int e     = desc >> 24;
    int slot0 = desc & 0xFFFFFF;
    int tid   = threadIdx.x;

    if (tid < TILE_M) {
        int nValid = cnt[e] - slot0;
        int s   = slot0 + ((tid < nValid) ? tid : 0);   // pad rows replicate slot0
        tokS[tid] = tokList[e * TK_T + s];
        gS[tid]   = (tid < nValid) ? gateList[e * TK_T + s] : 0.0f;
    }
    __syncthreads();

    // ---- stage gathered X rows into LDS with async-to-LDS b128 copies ----
    {
        // 48 rows x 512 bf16 = 3072 16B chunks; 12 per thread
#pragma unroll
        for (int j = 0; j < 12; ++j) {
            int c    = tid + j * 256;
            int row  = c >> 6;              // 64 chunks per row
            int col8 = (c & 63) * 8;
            unsigned long long gaddr =
                (unsigned long long)(xbf + (size_t)tokS[row] * TK_D + col8);
            unsigned ldsOff = (unsigned)(size_t)(&Xs[row * XS_STRIDE + col8]);
            asm volatile("global_load_async_to_lds_b128 %0, %1, off"
                         :: "v"(ldsOff), "v"(gaddr)
                         : "memory");
        }
        asm volatile("s_wait_asynccnt 0x0" ::: "memory");
    }
    __syncthreads();

    int wave  = tid >> 5;     // 0..7
    int lane  = tid & 31;
    int mrow  = lane & 15;    // row (A) / column (B,C,D) index within a tile
    int khalf = lane >> 4;    // which K half this lane holds

    const unsigned short* W1e = w1t + (size_t)e * TK_H * TK_D;
    const unsigned short* W2e = w2t + (size_t)e * TK_D * TK_H;

    v8f accY[3][4];           // [m-tile][d n-tile] persistent GEMM2 accumulators
#pragma unroll
    for (int m = 0; m < 3; ++m)
#pragma unroll
        for (int i = 0; i < 4; ++i) accY[m][i] = (v8f)0.0f;

    for (int hc = 0; hc < 8; ++hc) {        // H chunks of 256
        const int hbase = hc * 256;

        // ---- GEMM1: chunk columns [hbase, hbase+256); wave covers 32 cols ----
        {
            v8f a[3][2];                    // [m-tile][n-tile]
#pragma unroll
            for (int m = 0; m < 3; ++m)
#pragma unroll
                for (int nt = 0; nt < 2; ++nt) a[m][nt] = (v8f)0.0f;

            const unsigned short* Bbase =
                W1e + (size_t)(hbase + wave * 32 + mrow) * TK_D;

            for (int k = 0; k < TK_D; k += 32) {
                v16bf af[3];
#pragma unroll
                for (int m = 0; m < 3; ++m) {
                    const unsigned short* xr = &Xs[(mrow + 16 * m) * XS_STRIDE];
                    v8bf lo = *(const v8bf*)(xr + k + khalf * 16);
                    v8bf hi = *(const v8bf*)(xr + k + khalf * 16 + 8);
                    af[m] = __builtin_shufflevector(lo, hi,
                        0,1,2,3,4,5,6,7,8,9,10,11,12,13,14,15);
                }
#pragma unroll
                for (int nt = 0; nt < 2; ++nt) {
                    v16bf bfrag = *(const v16bf*)(Bbase + (size_t)nt * 16 * TK_D
                                                  + k + khalf * 16);
#pragma unroll
                    for (int m = 0; m < 3; ++m)
                        a[m][nt] = __builtin_amdgcn_wmma_f32_16x16x32_bf16(
                            false, af[m], false, bfrag, (short)0, a[m][nt],
                            false, false);
                }
            }

            // bias + relu + pack to bf16 in LDS (chunk-local column)
#pragma unroll
            for (int nt = 0; nt < 2; ++nt) {
                int hcol   = hbase + wave * 32 + nt * 16;
                float bias = b1[e * TK_H + hcol + mrow];
                int hlocal = wave * 32 + nt * 16 + mrow;
#pragma unroll
                for (int m = 0; m < 3; ++m) {
#pragma unroll
                    for (int r = 0; r < 8; ++r) {
                        float v = a[m][nt][r] + bias;
                        v = v > 0.0f ? v : 0.0f;
                        Hs[(m * 16 + khalf * 8 + r) * HS_STRIDE + hlocal] =
                            f2bf_raw(v);
                    }
                }
            }
        }
        __syncthreads();

        // ---- GEMM2 partial: K = this chunk's 256 h values ----
        {
            const unsigned short* Bbase =
                W2e + (size_t)(wave * 64 + mrow) * TK_H + hbase;

            for (int k = 0; k < 256; k += 32) {
                v16bf af[3];
#pragma unroll
                for (int m = 0; m < 3; ++m) {
                    const unsigned short* hr = &Hs[(mrow + 16 * m) * HS_STRIDE];
                    v8bf lo = *(const v8bf*)(hr + k + khalf * 16);
                    v8bf hi = *(const v8bf*)(hr + k + khalf * 16 + 8);
                    af[m] = __builtin_shufflevector(lo, hi,
                        0,1,2,3,4,5,6,7,8,9,10,11,12,13,14,15);
                }
#pragma unroll
                for (int nt = 0; nt < 4; ++nt) {
                    v16bf bfrag = *(const v16bf*)(Bbase + (size_t)nt * 16 * TK_H
                                                  + k + khalf * 16);
#pragma unroll
                    for (int m = 0; m < 3; ++m)
                        accY[m][nt] = __builtin_amdgcn_wmma_f32_16x16x32_bf16(
                            false, af[m], false, bfrag, (short)0, accY[m][nt],
                            false, false);
                }
            }
        }
        __syncthreads();
    }

    // ---- epilogue: out[token] += gate * (y + b2) ----
#pragma unroll
    for (int nt = 0; nt < 4; ++nt) {
        int dcol   = wave * 64 + nt * 16 + mrow;
        float bias = b2[e * TK_D + dcol];
#pragma unroll
        for (int m = 0; m < 3; ++m) {
#pragma unroll
            for (int r = 0; r < 8; ++r) {
                int row = m * 16 + khalf * 8 + r;
                float g = gS[row];
                if (g != 0.0f) {
                    float val = (accY[m][nt][r] + bias) * g;
                    atomicAdd(&out[(size_t)tokS[row] * TK_D + dcol], val);
                }
            }
        }
    }
}

// ---------------------------------------------------------------------------
// launch
// ---------------------------------------------------------------------------
extern "C" void kernel_launch(void* const* d_in, const int* in_sizes, int n_in,
                              void* d_out, int out_size, void* d_ws, size_t ws_size,
                              hipStream_t stream) {
    const float* x  = (const float*)d_in[0];
    const float* Wr = (const float*)d_in[1];
    const float* br = (const float*)d_in[2];
    const float* Wn = (const float*)d_in[3];
    const float* bn = (const float*)d_in[4];
    const float* W1 = (const float*)d_in[5];
    const float* b1 = (const float*)d_in[6];
    const float* W2 = (const float*)d_in[7];
    const float* b2 = (const float*)d_in[8];
    float* out = (float*)d_out;

    // workspace layout (all offsets 256B aligned)
    char* ws = (char*)d_ws;
    size_t off = 0;
    unsigned short* xbf = (unsigned short*)(ws + off); off += (size_t)TK_T * TK_D * 2;          // 16 MB
    unsigned short* w1t = (unsigned short*)(ws + off); off += (size_t)TK_E * TK_H * TK_D * 2;   // 16 MB
    unsigned short* w2t = (unsigned short*)(ws + off); off += (size_t)TK_E * TK_D * TK_H * 2;   // 16 MB
    int*   cnt      = (int*)(ws + off);   off += 256;
    int*   tokList  = (int*)(ws + off);   off += (size_t)TK_E * TK_T * 4;                       // 512 KB
    float* gateList = (float*)(ws + off); off += (size_t)TK_E * TK_T * 4;                       // 512 KB
    int*   tileDesc = (int*)(ws + off);   off += 4096 * 4;
    int*   numTiles = (int*)(ws + off);   off += 256;
    (void)ws_size; (void)n_in; (void)in_sizes; (void)out_size;

    // allow >64KB dynamic LDS for the expert kernel (CDNA5: 320KB per WGP)
    static bool attrDone = false;
    if (!attrDone) {
        (void)hipFuncSetAttribute((const void*)moe_expert_kernel,
                                  hipFuncAttributeMaxDynamicSharedMemorySize,
                                  SMEM_BYTES);
        attrDone = true;
    }

    const int nOut = TK_T * TK_D;

    moe_zero_kernel<<<1024, 256, 0, stream>>>(out, cnt, nOut);
    moe_cvt_x_kernel<<<2048, 256, 0, stream>>>(x, xbf, TK_T * TK_D);
    moe_tw1_kernel<<<2048, 256, 0, stream>>>(W1, w1t, TK_E * TK_H * TK_D);
    moe_tw2_kernel<<<2048, 256, 0, stream>>>(W2, w2t, TK_E * TK_D * TK_H);
    moe_router_kernel<<<TK_T / 256, 256, 0, stream>>>(x, Wr, br, Wn, bn,
                                                      cnt, tokList, gateList);
    moe_sched_kernel<<<1, 64, 0, stream>>>(cnt, tileDesc, numTiles);
    moe_expert_kernel<<<EXPERT_GRID, 256, SMEM_BYTES, stream>>>(
        xbf, w1t, w2t, b1, b2, tokList, gateList, cnt, tileDesc, numTiles, out);
}
